// FourierBlock_71021579207178
// MI455X (gfx1250) — matrix operand order repaired and checked by reading, not compiled, not run on hardware
//
#include <hip/hip_runtime.h>
#include <hip/hip_bf16.h>

// FourierBlock: truncated-rfft -> per-(k,h) complex 64x64 mix -> truncated-irfft.
// Implemented as three f16 WMMA GEMM stages with f32 accumulation.
// B=8, L=4096, H=8, EIN=EOUT=64, K=64 kept frequencies, C = H*EIN = 512.

#define LFFT 4096
#define KFREQ 64
#define NBATCH 8
#define CCH 512          // H*EOUT = H*EIN
#define MROWS 128        // 64 cos rows + 64 (-sin) rows

typedef _Float16 h16 __attribute__((ext_vector_type(16)));
typedef _Float16 h8  __attribute__((ext_vector_type(8)));
typedef float    f8  __attribute__((ext_vector_type(8)));

__device__ __forceinline__ f8 wmma_f16(h16 a, h16 b, f8 c) {
    // D = A(16x32 f16) * B(32x16 f16) + C(16x16 f32)
    return __builtin_amdgcn_wmma_f32_16x16x32_f16(
        /*neg_a=*/false, a, /*neg_b=*/false, b,
        /*c_mod=*/(short)0, c, /*reuse_a=*/false, /*reuse_b=*/false);
}

// ---- tile loaders (ISA 7.12.2 layouts, wave32) --------------------------
// A 16x32 f16: lanes 0-15 hold row M=lane, K=0..7 (v0-3) and 16..23 (v4-7);
//              lanes 16-31 hold row M=lane-16, K=8..15 and 24..31.
__device__ __forceinline__ h16 load_a_f16(const _Float16* p, int ld, int m0,
                                          int k0, int lane) {
    int m  = m0 + (lane & 15);
    int kb = k0 + ((lane & 16) ? 8 : 0);
    const _Float16* q0 = p + (size_t)m * ld + kb;
    h8 lo = *(const h8*)(q0);        // K = kb .. kb+7   (16B aligned)
    h8 hi = *(const h8*)(q0 + 16);   // K = kb+16 .. kb+23
    h16 r;
#pragma unroll
    for (int j = 0; j < 8; ++j) { r[j] = lo[j]; r[8 + j] = hi[j]; }
    return r;
}

// B 32x16 f16: lanes 0-15 hold column N=lane, K=0..7 and 16..23; lanes 16-31
//              column N=lane-16, K=8..15 and 24..31 (two halves per VGPR).
__device__ __forceinline__ h16 load_b_f32cvt(const float* p, int ld, int k0,
                                             int n0, int lane, float scale) {
    int n  = n0 + (lane & 15);
    int kb = k0 + ((lane & 16) ? 8 : 0);
    h16 r;
#pragma unroll
    for (int j = 0; j < 16; ++j)
        r[j] = (_Float16)(p[(size_t)(kb + j + (j & 8)) * ld + n] * scale);
    return r;
}

__device__ __forceinline__ h16 load_b_f16(const _Float16* p, int ld, int k0,
                                          int n0, int lane) {
    int n  = n0 + (lane & 15);
    int kb = k0 + ((lane & 16) ? 8 : 0);
    h16 r;
#pragma unroll
    for (int j = 0; j < 16; ++j) r[j] = p[(size_t)(kb + j + (j & 8)) * ld + n];
    return r;
}

// D 16x16 f32: vgpr v -> row (v + 8*(lane>=16)), col = lane&15.
__device__ __forceinline__ void store_d_f16(_Float16* p, int ld, int m0,
                                            int n0, int lane, f8 acc) {
    int n  = n0 + (lane & 15);
    int r0 = m0 + ((lane & 16) ? 8 : 0);
#pragma unroll
    for (int v = 0; v < 8; ++v) p[(size_t)(r0 + v) * ld + n] = (_Float16)acc[v];
}

// ---- kernel 0: twiddle tables -------------------------------------------
// Tf  [128][4096]: m<64 -> cos(2pi*m*t/L), m>=64 -> -sin(2pi*(m-64)*t/L)
// Tinv[4096][128]: kk<64 -> cos(2pi*kk*t/L) (kk==0 weighted 0.5),
//                  kk>=64 -> -sin(2pi*(kk-64)*t/L)
__global__ void init_twiddles(_Float16* Tf, _Float16* Tinv) {
    int idx = blockIdx.x * blockDim.x + threadIdx.x;
    if (idx >= MROWS * LFFT) return;
    const float w = 6.28318530717958647692f / (float)LFFT;
    {   // Tf[m][t]
        int m = idx / LFFT, t = idx % LFFT;
        int k = (m < KFREQ) ? m : (m - KFREQ);
        float th = w * (float)((k * t) & (LFFT - 1));
        float v = (m < KFREQ) ? __cosf(th) : -__sinf(th);
        Tf[idx] = (_Float16)v;
    }
    {   // Tinv[t][kk]
        int t = idx / MROWS, kk = idx % MROWS;
        int k = (kk < KFREQ) ? kk : (kk - KFREQ);
        float th = w * (float)((k * t) & (LFFT - 1));
        float v;
        if (kk < KFREQ) { v = __cosf(th); if (kk == 0) v = 0.5f; }
        else            { v = -__sinf(th); }
        Tinv[idx] = (_Float16)v;
    }
}

// ---- kernel 1: forward DFT  X[b] = Tf(128x4096) @ q[b](4096x512) --------
// One wave per 32(M) x 32(N) tile.  grid = (4, 16, 8)
__global__ void __launch_bounds__(32) fwd_dft(const float* __restrict__ q,
                                              const _Float16* __restrict__ Tf,
                                              _Float16* __restrict__ Xbuf) {
    int lane = threadIdx.x;
    int m0 = blockIdx.x * 32;
    int n0 = blockIdx.y * 32;
    int b  = blockIdx.z;
    const float* qb = q + (size_t)b * LFFT * CCH;
    f8 acc[2][2] = {};
    for (int k0 = 0; k0 < LFFT; k0 += 32) {
        h16 a0 = load_a_f16(Tf, LFFT, m0,      k0, lane);
        h16 a1 = load_a_f16(Tf, LFFT, m0 + 16, k0, lane);
#pragma unroll
        for (int j = 0; j < 2; ++j) {
            h16 bt = load_b_f32cvt(qb, CCH, k0, n0 + 16 * j, lane, 1.0f);
            acc[0][j] = wmma_f16(a0, bt, acc[0][j]);
            acc[1][j] = wmma_f16(a1, bt, acc[1][j]);
        }
    }
    _Float16* Xb = Xbuf + (size_t)b * MROWS * CCH;
#pragma unroll
    for (int i = 0; i < 2; ++i)
#pragma unroll
        for (int j = 0; j < 2; ++j)
            store_d_f16(Xb, CCH, m0 + 16 * i, n0 + 16 * j, lane, acc[i][j]);
}

// ---- kernel 2: per-(k,h) complex mix ------------------------------------
// A rows = batch b (8 valid of 16).  W scaled by 2^16 to stay f16-normal.
// out_re = Xre*Wre + Xim*(-Wim) ; out_im = Xre*Wim + Xim*Wre
__global__ void __launch_bounds__(32) cplx_mix(const float* __restrict__ Wre,
                                               const float* __restrict__ Wim,
                                               const _Float16* __restrict__ Xbuf,
                                               _Float16* __restrict__ Obuf) {
    int lane = threadIdx.x;
    int kh = blockIdx.x;           // k*8 + h
    int k = kh >> 3, h = kh & 7;
    int m  = lane & 15;            // row = batch index
    int kb = (lane & 16) ? 8 : 0;  // A-layout K split
    h16 are[2], aim[2];
#pragma unroll
    for (int kc = 0; kc < 2; ++kc) {
        if (m < NBATCH) {
            const _Float16* pr = Xbuf + ((size_t)m * MROWS + k) * CCH
                                 + h * 64 + kc * 32 + kb;
            const _Float16* pi = Xbuf + ((size_t)m * MROWS + KFREQ + k) * CCH
                                 + h * 64 + kc * 32 + kb;
            h8 rl = *(const h8*)(pr); h8 rh = *(const h8*)(pr + 16);
            h8 il = *(const h8*)(pi); h8 ih = *(const h8*)(pi + 16);
#pragma unroll
            for (int j = 0; j < 8; ++j) {
                are[kc][j] = rl[j]; are[kc][8 + j] = rh[j];
                aim[kc][j] = il[j]; aim[kc][8 + j] = ih[j];
            }
        } else {
#pragma unroll
            for (int j = 0; j < 16; ++j) {
                are[kc][j] = (_Float16)0.0f; aim[kc][j] = (_Float16)0.0f;
            }
        }
    }
    const float* wr = Wre + (size_t)kh * 64 * 64;
    const float* wi = Wim + (size_t)kh * 64 * 64;
    const float WS = 65536.0f;     // 2^16, removed in final scale
#pragma unroll
    for (int jt = 0; jt < 4; ++jt) {
        int n0 = jt * 16;
        f8 ar = {}, ai = {};
#pragma unroll
        for (int kc = 0; kc < 2; ++kc) {
            h16 bre = load_b_f32cvt(wr, 64, kc * 32, n0, lane, WS);
            h16 bim = load_b_f32cvt(wi, 64, kc * 32, n0, lane, WS);
            h16 bimn;
#pragma unroll
            for (int j = 0; j < 16; ++j) bimn[j] = -bim[j];
            ar = wmma_f16(are[kc], bre,  ar);
            ar = wmma_f16(aim[kc], bimn, ar);
            ai = wmma_f16(are[kc], bim,  ai);
            ai = wmma_f16(aim[kc], bre,  ai);
        }
        if (!(lane & 16)) {        // rows 0..7 == valid batches
            int c = h * 64 + n0 + (lane & 15);
#pragma unroll
            for (int v = 0; v < 8; ++v) {
                Obuf[((size_t)v * MROWS + k) * CCH + c]         = (_Float16)ar[v];
                Obuf[((size_t)v * MROWS + KFREQ + k) * CCH + c] = (_Float16)ai[v];
            }
        }
    }
}

// ---- kernel 3: inverse  y[b] = Tinv(4096x128) @ O[b](128x512) * 2^-27 ---
// One wave per 32(M=t) x 64(N) tile.  grid = (128, 8, 8)
__global__ void __launch_bounds__(32) inv_dft(const _Float16* __restrict__ Tinv,
                                              const _Float16* __restrict__ Obuf,
                                              float* __restrict__ y) {
    int lane = threadIdx.x;
    int m0 = blockIdx.x * 32;
    int n0 = blockIdx.y * 64;
    int b  = blockIdx.z;
    const _Float16* Ob = Obuf + (size_t)b * MROWS * CCH;
    f8 acc[2][4] = {};
#pragma unroll
    for (int k0 = 0; k0 < MROWS; k0 += 32) {
        h16 a0 = load_a_f16(Tinv, MROWS, m0,      k0, lane);
        h16 a1 = load_a_f16(Tinv, MROWS, m0 + 16, k0, lane);
#pragma unroll
        for (int j = 0; j < 4; ++j) {
            h16 bt = load_b_f16(Ob, CCH, k0, n0 + 16 * j, lane);
            acc[0][j] = wmma_f16(a0, bt, acc[0][j]);
            acc[1][j] = wmma_f16(a1, bt, acc[1][j]);
        }
    }
    float* yb = y + (size_t)b * LFFT * CCH;
    const float scale = 7.450580596923828e-09f;   // 2/(L * 2^16) = 2^-27
#pragma unroll
    for (int i = 0; i < 2; ++i)
#pragma unroll
        for (int j = 0; j < 4; ++j) {
            int n  = n0 + 16 * j + (lane & 15);
            int r0 = m0 + 16 * i + ((lane & 16) ? 8 : 0);
#pragma unroll
            for (int v = 0; v < 8; ++v)
                yb[(size_t)(r0 + v) * CCH + n] = acc[i][j][v] * scale;
        }
}

extern "C" void kernel_launch(void* const* d_in, const int* in_sizes, int n_in,
                              void* d_out, int out_size, void* d_ws, size_t ws_size,
                              hipStream_t stream) {
    // dict order: q, k, v, W_real, W_imag, mask
    const float* q   = (const float*)d_in[0];
    const float* Wre = (const float*)d_in[3];
    const float* Wim = (const float*)d_in[4];
    float* y = (float*)d_out;

    // workspace (f16): Tf | Tinv | Xbuf | Obuf  -> 4 x 1 MB = 4 MB total
    _Float16* Tf   = (_Float16*)d_ws;
    _Float16* Tinv = Tf   + (size_t)MROWS * LFFT;
    _Float16* Xbuf = Tinv + (size_t)MROWS * LFFT;
    _Float16* Obuf = Xbuf + (size_t)NBATCH * MROWS * CCH;

    int telems = MROWS * LFFT;
    init_twiddles<<<(telems + 255) / 256, 256, 0, stream>>>(Tf, Tinv);
    fwd_dft<<<dim3(4, 16, NBATCH), 32, 0, stream>>>(q, Tf, Xbuf);
    cplx_mix<<<KFREQ * 8, 32, 0, stream>>>(Wre, Wim, Xbuf, Obuf);
    inv_dft<<<dim3(LFFT / 32, CCH / 64, NBATCH), 32, 0, stream>>>(Tinv, Obuf, y);
}
